// LossObj3_11879879542627
// MI455X (gfx1250) — matrix operand order repaired and checked by reading, not compiled, not run on hardware
//
#include <hip/hip_runtime.h>
#include <hip/hip_bf16.h>
#include <stdint.h>

// Problem constants (from reference): logits [4,19,512,1024], label [4,512,1024]
#define BB    4
#define CC    19
#define HH    512
#define WW    1024
#define HW    (HH * WW)            // 524288
#define NTOT  (BB * HW)            // 2097152 pixels
#define IGNORE_LBL 255

#define TILE_PIX   512             // pixels per build-keys block
#define TILES_PER_IMG (HW / TILE_PIX)   // 1024
#define RADIX_ELEMS 16
#define BLK_ELEMS   (256 * RADIX_ELEMS) // 4096
#define NBLK        (NTOT / BLK_ELEMS)  // 512

typedef _Float16 v16h __attribute__((ext_vector_type(16)));
typedef float    v8f  __attribute__((ext_vector_type(8)));
typedef unsigned int u32x4 __attribute__((ext_vector_type(4)));
typedef int          i32x8 __attribute__((ext_vector_type(8)));
typedef int          i32x4 __attribute__((ext_vector_type(4)));

// ---------------------------------------------------------------------------
// Kernel 1: softmax + per-class error keys.
// key k = (float_bits(err) << 1) | fg  (err in [0,1] -> fits in 31 bits)
// stored as ~k so that ASCENDING radix sort == DESCENDING err, fg=1 first in ties.
// Logits tile (19 rows x 512 pixels, row stride HW) is staged into LDS via the
// Tensor Data Mover (CDNA5 TDM) when available.
// ---------------------------------------------------------------------------
__global__ void build_keys_kernel(const float* __restrict__ logits,
                                  const int* __restrict__ label,
                                  uint32_t* __restrict__ keys) {
  __shared__ float lbuf[CC * TILE_PIX];          // 19*512*4 = 38 KB
  const int t = threadIdx.x;
  const int g = blockIdx.x;                      // 0..4095
  const int b = g / TILES_PER_IMG;
  const int pix0 = (g % TILES_PER_IMG) * TILE_PIX;
  const long long gelem = (long long)b * CC * HW + pix0;  // element offset of tile row 0

#if defined(__gfx1250__) && __has_builtin(__builtin_amdgcn_tensor_load_to_lds) && \
    __has_builtin(__builtin_amdgcn_s_wait_tensorcnt)
  if (t == 0) {
    unsigned long long ga =
        (unsigned long long)(uintptr_t)logits + (unsigned long long)gelem * 4ull;
    uint32_t lds0 = (uint32_t)(uintptr_t)&lbuf[0];
    // D# group0: count=1 | lds_addr | global_addr[56:0] | type=2
    u32x4 g0 = { 1u, lds0, (uint32_t)ga,
                 (uint32_t)((ga >> 32) & 0x1FFFFFFull) | (2u << 30) };
    // D# group1 (2D tile): data_size=4B (code 2); tensor_dim0=HW; tensor_dim1=B*C;
    // tile_dim0=TILE_PIX; tile_dim1=CC; tensor_dim0_stride=HW
    i32x8 g1 = {
      (int)0x00020000,                       // d0: wg_mask=0, data_size=2
      (int)0x00000000,                       // d1: bar_addr=0, tensor_dim0 lo16=0
      (int)(((BB * CC) << 16) | (HW >> 16)), // d2: tensor_dim0 hi16 | tensor_dim1 lo16
      (int)(TILE_PIX << 16),                 // d3: tensor_dim1 hi16=0 | tile_dim0
      (int)CC,                               // d4: tile_dim1=19, tile_dim2=0
      (int)HW,                               // d5: tensor_dim0_stride lo32
      0, 0                                   // d6,d7: stride hi / dim1_stride = 0
    };
    i32x4 g2 = {0, 0, 0, 0};
    i32x4 g3 = {0, 0, 0, 0};
    i32x8 g4 = {0, 0, 0, 0, 0, 0, 0, 0};
    __builtin_amdgcn_tensor_load_to_lds(g0, g1, g2, g3, g4, 0);
    __builtin_amdgcn_s_wait_tensorcnt(0);
  }
  __syncthreads();
#else
  for (int c = 0; c < CC; ++c)
    for (int p = t; p < TILE_PIX; p += 256)
      lbuf[c * TILE_PIX + p] = logits[gelem + (long long)c * HW + p];
  __syncthreads();
#endif

  for (int pp = t; pp < TILE_PIX; pp += 256) {
    const long long n = (long long)b * HW + pix0 + pp;
    const int lab = label[n];
    const bool valid = (lab != IGNORE_LBL);
    float m = -1e30f;
    for (int c = 0; c < CC; ++c) m = fmaxf(m, lbuf[c * TILE_PIX + pp]);
    float sum = 0.f;
    for (int c = 0; c < CC; ++c) sum += __expf(lbuf[c * TILE_PIX + pp] - m);
    const float inv = 1.f / sum;
    for (int c = 0; c < CC; ++c) {
      const float pc = __expf(lbuf[c * TILE_PIX + pp] - m) * inv;
      const uint32_t fg = (valid && lab == c) ? 1u : 0u;
      const float err = valid ? fabsf((float)fg - pc) : 0.f;
      const uint32_t k = (__float_as_uint(err) << 1) | fg;
      keys[(size_t)c * NTOT + n] = ~k;
    }
  }
}

// ---------------------------------------------------------------------------
// Radix sort pass kernels (stable LSD, 8-bit digits, ascending).
// grid = (NBLK, 19); 256 threads; 16 elems/thread, strided ownership.
// ---------------------------------------------------------------------------
__global__ void radix_hist_kernel(const uint32_t* __restrict__ src,
                                  uint32_t* __restrict__ ghist, int shift) {
  const int cls = blockIdx.y, blk = blockIdx.x, t = threadIdx.x;
  __shared__ uint32_t h[256];
  h[t] = 0;
  __syncthreads();
  const uint32_t* p = src + (size_t)cls * NTOT + (size_t)blk * BLK_ELEMS;
  for (int r = 0; r < RADIX_ELEMS; ++r)
    atomicAdd(&h[(p[r * 256 + t] >> shift) & 255u], 1u);
  __syncthreads();
  ghist[((size_t)cls * 256 + t) * NBLK + blk] = h[t];
}

__global__ void radix_scan_kernel(uint32_t* __restrict__ ghist) {
  // exclusive scan over 256*NBLK counters of one class (digit-major)
  const int cls = blockIdx.x, t = threadIdx.x;
  uint32_t* base = ghist + (size_t)cls * 256 * NBLK;
  const int CH = (256 * NBLK) / 256;  // 512
  __shared__ uint32_t sums[256];
  __shared__ uint32_t excl[256];
  uint32_t s = 0;
  for (int j = 0; j < CH; ++j) s += base[t * CH + j];
  sums[t] = s;
  __syncthreads();
  if (t == 0) {
    uint32_t run = 0;
    for (int i = 0; i < 256; ++i) { excl[i] = run; run += sums[i]; }
  }
  __syncthreads();
  uint32_t run = excl[t];
  for (int j = 0; j < CH; ++j) {
    uint32_t v = base[t * CH + j];
    base[t * CH + j] = run;
    run += v;
  }
}

__global__ void radix_scatter_kernel(const uint32_t* __restrict__ src,
                                     uint32_t* __restrict__ dst,
                                     const uint32_t* __restrict__ ghist, int shift) {
  const int cls = blockIdx.y, blk = blockIdx.x, t = threadIdx.x;
  const int lane = t & 31, w = t >> 5;
  __shared__ uint32_t whist[8 * 256];
  __shared__ uint32_t rtot[256];
  __shared__ uint32_t bbase[256];
  bbase[t] = 0;
  const uint32_t* p = src + (size_t)cls * NTOT + (size_t)blk * BLK_ELEMS;
  uint32_t keys[RADIX_ELEMS];
  for (int r = 0; r < RADIX_ELEMS; ++r) keys[r] = p[r * 256 + t];
  __syncthreads();

  for (int r = 0; r < RADIX_ELEMS; ++r) {
    for (int i = t; i < 2048; i += 256) whist[i] = 0;
    __syncthreads();
    const uint32_t k = keys[r];
    const uint32_t d = (k >> shift) & 255u;
    // wave32 digit-match mask via ballots
    uint32_t m = 0xffffffffu;
    for (int bb = 0; bb < 8; ++bb) {
      uint32_t bit = (d >> bb) & 1u;
      uint32_t bal = (uint32_t)__ballot((int)bit);
      m &= bit ? bal : ~bal;
    }
    const uint32_t before = __popc(m & ((1u << lane) - 1u));
    if (before == 0) whist[w * 256 + d] = __popc(m);   // group leader
    __syncthreads();
    {   // per-digit exclusive scan across the 8 waves (thread t <-> digit t)
      uint32_t run = 0;
      for (int ww = 0; ww < 8; ++ww) {
        uint32_t v = whist[ww * 256 + t];
        whist[ww * 256 + t] = run;
        run += v;
      }
      rtot[t] = run;
    }
    __syncthreads();
    const uint32_t rank = bbase[d] + whist[w * 256 + d] + before;
    const uint32_t pos = ghist[((size_t)cls * 256 + d) * NBLK + blk] + rank;
    dst[(size_t)cls * NTOT + pos] = k;
    __syncthreads();
    bbase[t] += rtot[t];
    __syncthreads();
  }
}

// ---------------------------------------------------------------------------
// fg prefix machinery over sorted keys
// ---------------------------------------------------------------------------
__global__ void fg_blocksum_kernel(const uint32_t* __restrict__ keys,
                                   uint32_t* __restrict__ blockFg) {
  const int cls = blockIdx.y, blk = blockIdx.x, t = threadIdx.x;
  __shared__ uint32_t red[256];
  const uint32_t* p = keys + (size_t)cls * NTOT + (size_t)blk * BLK_ELEMS;
  uint32_t s = 0;
  for (int r = 0; r < RADIX_ELEMS; ++r) s += (p[r * 256 + t] & 1u) ^ 1u;  // fg = ~key & 1
  red[t] = s;
  __syncthreads();
  for (int o = 128; o > 0; o >>= 1) {
    if (t < o) red[t] += red[t + o];
    __syncthreads();
  }
  if (t == 0) blockFg[(size_t)cls * NBLK + blk] = red[0];
}

__global__ void fg_scan_kernel(uint32_t* __restrict__ blockFg,
                               uint32_t* __restrict__ classG) {
  const int cls = blockIdx.x;
  if (threadIdx.x == 0) {
    uint32_t run = 0;
    uint32_t* bp = blockFg + (size_t)cls * NBLK;
    for (int i = 0; i < NBLK; ++i) { uint32_t v = bp[i]; bp[i] = run; run += v; }
    classG[cls] = run;   // G = total fg count for class
  }
}

// ---------------------------------------------------------------------------
// Final: loss_c = sum_i jacc_i * (e_i - e_{i+1}),
//   jacc_i = 1 - (G - F_i) / (G + (i+1) - F_i)
// Cross-thread exclusive scan of per-thread fg counts done with
// v_wmma_f32_16x16x32_f16:  W = S(16x16) x U(upper-tri ones), exact in f16/f32.
// ---------------------------------------------------------------------------
__global__ void lovasz_final_kernel(const uint32_t* __restrict__ keys,
                                    const uint32_t* __restrict__ blockFgExcl,
                                    const uint32_t* __restrict__ classG,
                                    float* __restrict__ out) {
  const int cls = blockIdx.y, blk = blockIdx.x, t = threadIdx.x;
  const uint32_t* p = keys + (size_t)cls * NTOT + (size_t)blk * BLK_ELEMS;
  const int base = t * RADIX_ELEMS;                  // contiguous chunk per thread
  uint32_t myk[RADIX_ELEMS];
  for (int e = 0; e < RADIX_ELEMS; ++e) myk[e] = p[base + e];

  int s = 0;
  for (int e = 0; e < RADIX_ELEMS; ++e) s += (int)((myk[e] & 1u) ^ 1u);

  __shared__ _Float16 lds_s[256];
  __shared__ float lds_w[256];
  __shared__ float lds_rx[16];
  __shared__ float red[256];
  lds_s[t] = (_Float16)(float)s;
  __syncthreads();

  if (t < 32) {
    // A (16x32 f16): row M = t&15; lanes 0-15 hold K=0..7, lanes 16-31 K=8..15;
    // elements 8..15 (K=16..31) are zero padding.
    const int M = t & 15;
    const int c0 = (t < 16) ? 0 : 8;
    v16h a, bm;
    for (int j = 0; j < 8; ++j) a[j] = lds_s[M * 16 + c0 + j];
    for (int j = 8; j < 16; ++j) a[j] = (_Float16)0.0f;
    // B (32x16 f16) = upper-triangular ones in K=0..15; column N = lane%16; lanes 0-15
    // hold K=0..7, lanes 16-31 K=8..15; elements 8..15 (K=16..31) are zero.
    const int Nn = t & 15;
    for (int kk = 0; kk < 16; ++kk)
      bm[kk] = (kk < 8 && c0 + kk <= Nn) ? (_Float16)1.0f : (_Float16)0.0f;
    v8f c;
    for (int j = 0; j < 8; ++j) c[j] = 0.f;
    c = __builtin_amdgcn_wmma_f32_16x16x32_f16(false, a, false, bm,
                                               (short)0, c, false, false);
    // D layout: VGPR j -> M = j (+8 for lanes 16-31), N = lane%16
    const int Mb = (t < 16) ? 0 : 8;
    for (int j = 0; j < 8; ++j) lds_w[(j + Mb) * 16 + Nn] = c[j];
  }
  __syncthreads();
  if (t == 0) {           // exclusive scan of the 16 row totals
    float run = 0.f;
    for (int r = 0; r < 16; ++r) { lds_rx[r] = run; run += lds_w[r * 16 + 15]; }
  }
  __syncthreads();

  const int rr = t >> 4, col = t & 15;
  const float exclF = lds_rx[rr] + lds_w[rr * 16 + col] - (float)s; // excl over threads
  const float Fbase = (float)blockFgExcl[(size_t)cls * NBLK + blk] + exclF;
  const float G = (float)classG[cls];

  float acc = 0.f;
  int fgrun = 0;
  const long long i0 = (long long)blk * BLK_ELEMS + base;
  for (int e = 0; e < RADIX_ELEMS; ++e) {
    const uint32_t k = ~myk[e];
    fgrun += (int)(k & 1u);
    const float err = __uint_as_float(k >> 1);
    const long long i = i0 + e;
    float enext = 0.f;
    if (i + 1 < (long long)NTOT)
      enext = __uint_as_float((~keys[(size_t)cls * NTOT + i + 1]) >> 1);
    const float Fi = Fbase + (float)fgrun;          // inclusive fg count through i
    const float uni = G + (float)(i + 1) - Fi;      // union >= i+1 >= 1
    const float jac = 1.f - (G - Fi) / uni;
    acc += jac * (err - enext);
  }

  red[t] = acc;
  __syncthreads();
  for (int o = 128; o > 0; o >>= 1) {
    if (t < o) red[t] += red[t + o];
    __syncthreads();
  }
  if (t == 0) atomicAdd(out, red[0]);
}

// ---------------------------------------------------------------------------
extern "C" void kernel_launch(void* const* d_in, const int* in_sizes, int n_in,
                              void* d_out, int out_size, void* d_ws, size_t ws_size,
                              hipStream_t stream) {
  (void)in_sizes; (void)n_in; (void)out_size; (void)ws_size;
  const float* logits = (const float*)d_in[0];
  const int* label = (const int*)d_in[1];
  float* out = (float*)d_out;

  // workspace layout (~314 MB): keysA | keysB | ghist | blockFg | classG
  uint32_t* keysA   = (uint32_t*)d_ws;
  uint32_t* keysB   = keysA + (size_t)CC * NTOT;
  uint32_t* ghist   = keysB + (size_t)CC * NTOT;
  uint32_t* blockFg = ghist + (size_t)CC * 256 * NBLK;
  uint32_t* classG  = blockFg + (size_t)CC * NBLK;

  (void)hipMemsetAsync(d_out, 0, sizeof(float), stream);

  build_keys_kernel<<<BB * TILES_PER_IMG, 256, 0, stream>>>(logits, label, keysA);

  uint32_t* bufs[2] = {keysA, keysB};
  for (int pass = 0; pass < 4; ++pass) {
    uint32_t* src = bufs[pass & 1];
    uint32_t* dst = bufs[(pass & 1) ^ 1];
    const int shift = pass * 8;
    radix_hist_kernel<<<dim3(NBLK, CC), 256, 0, stream>>>(src, ghist, shift);
    radix_scan_kernel<<<CC, 256, 0, stream>>>(ghist);
    radix_scatter_kernel<<<dim3(NBLK, CC), 256, 0, stream>>>(src, dst, ghist, shift);
  }
  // after 4 passes sorted data is back in keysA (ascending ~k == descending err)

  fg_blocksum_kernel<<<dim3(NBLK, CC), 256, 0, stream>>>(keysA, blockFg);
  fg_scan_kernel<<<CC, 32, 0, stream>>>(blockFg, classG);
  lovasz_final_kernel<<<dim3(NBLK, CC), 256, 0, stream>>>(keysA, blockFg, classG, out);
}
